// LeapfrogIntegrator_31181462569462
// MI455X (gfx1250) — compile-verified, matrix operand order checked by reading, not run on hardware
//
#include <hip/hip_runtime.h>
#include <cmath>

// ---------------------------------------------------------------------------
// Hyperbolic (Poincare-ball) leapfrog integrator, fused single kernel.
// B=131072 points, D=64, H=512, 4 steps. Compute-bound on the MLP
// fwd+bwd GEMMs (137 GFLOP) -> v_wmma_f32_16x16x32_f16 with f32 accum.
// One point per lane; each wave32 owns 32 rows = two 16-row WMMA tiles.
// W1 is DMA'd into LDS with the Tensor Data Mover (tensor_load_to_lds),
// then pre-swizzled in LDS into per-lane contiguous WMMA B-fragments (both
// W1 for the forward K=D contraction and W1^T for the backward K=H one).
// ---------------------------------------------------------------------------

typedef __attribute__((ext_vector_type(16))) _Float16 v16h;
typedef __attribute__((ext_vector_type(8)))  _Float16 v8h;
typedef __attribute__((ext_vector_type(8)))  float    v8f;
typedef __attribute__((ext_vector_type(4)))  float    v4f;
typedef __attribute__((ext_vector_type(4)))  unsigned int u32x4;
typedef __attribute__((ext_vector_type(8)))  int      i32x8;
typedef __attribute__((ext_vector_type(4)))  int      i32x4;

#define HEPS   1e-5f
#define HDT    0.1f
#define HSTEPS 4
#define DDIM   64
#define HDIM   512
#define THREADS 256
#define WAVES   8

// LDS layout (bytes)
#define OFF_W1F 0        // 64 frags * 32 lanes * 32B  = 65536 (fwd B-frags)
#define OFF_W1B 65536    // 64 frags * 32 lanes * 32B  = 65536 (bwd B-frags)
#define OFF_B1  131072   // 512 f32
#define OFF_W2  133120   // 512 f32
#define OFF_STG 135168   // 8 waves * 8192B staging (also raw-W1 TDM target)
#define STG_PER_WAVE 8192
#define SMEM_BYTES (OFF_STG + WAVES * STG_PER_WAVE)   // 200704

#if __has_builtin(__builtin_amdgcn_tensor_load_to_lds) && \
    __has_builtin(__builtin_amdgcn_s_wait_tensorcnt)
#define HAVE_TDM 1
#endif

__device__ __forceinline__ float fast_tanh(float x) {
#if __has_builtin(__builtin_amdgcn_tanhf)
  return __builtin_amdgcn_tanhf(x);        // v_tanh_f32 (gfx1250 TRANS op)
#else
  float r;
  asm("v_tanh_f32 %0, %1" : "=v"(r) : "v"(x));
  return r;
#endif
}

#ifdef HAVE_TDM
// TDM: DMA a 32x512 f32 tile (64KB) from global into LDS at byte offset lds_off.
// D# group0: count=1 | lds_addr | global_addr[56:0] | type=2 ("image").
// D# group1: data_size=4B, tensor_dim0=512, tensor_dim1=32, tile 512x32,
//            tensor_dim0_stride=512 elements. Groups 2/3 unused (2D tensor).
__device__ __forceinline__ void tdm_load_w1_chunk(const float* gsrc, unsigned lds_off) {
  unsigned long long ga = (unsigned long long)(__SIZE_TYPE__)gsrc;
  u32x4 g0;
  g0[0] = 1u;                                               // count=1, user D#
  g0[1] = lds_off;                                          // lds_addr (bytes)
  g0[2] = (unsigned)(ga & 0xFFFFFFFFu);                     // global_addr[31:0]
  g0[3] = (unsigned)((ga >> 32) & 0x1FFFFFFu) | (2u << 30); // addr[56:32]|type=2
  i32x8 g1;
  g1[0] = (int)(2u << 16);          // workgroup_mask=0, data_size=2 (4 bytes)
  g1[1] = (int)(512u << 16);        // tensor_dim0 = 512 (bits 79:48, low half)
  g1[2] = (int)(32u << 16);         // tensor_dim0 hi=0 | tensor_dim1 = 32
  g1[3] = (int)(512u << 16);        // tensor_dim1 hi=0 | tile_dim0 = 512
  g1[4] = 32;                       // tile_dim1 = 32, tile_dim2 = 0
  g1[5] = 512;                      // tensor_dim0_stride lo (elements)
  g1[6] = 0;                        // stride hi | tensor_dim1_stride lo
  g1[7] = 0;                        // tensor_dim1_stride hi
  i32x4 z4 = {0, 0, 0, 0};
  i32x8 z8 = {0, 0, 0, 0, 0, 0, 0, 0};
  __builtin_amdgcn_tensor_load_to_lds(g0, g1, z4, z4, z8, 0);
}
#endif

// 16 contiguous f16 (one pre-swizzled fragment slab)
__device__ __forceinline__ v16h ld16c(const _Float16* p) {
  v8h lo = *(const v8h*)(p);
  v8h hi = *(const v8h*)(p + 8);
  v16h r;
#pragma unroll
  for (int i = 0; i < 8; ++i) { r[i] = lo[i]; r[i + 8] = hi[i]; }
  return r;
}

// A-fragment from a row-major staging row: 8 f16 at +0, 8 f16 at +16 elems
__device__ __forceinline__ v16h ld16s(const _Float16* p) {
  v8h lo = *(const v8h*)(p);
  v8h hi = *(const v8h*)(p + 16);
  v16h r;
#pragma unroll
  for (int i = 0; i < 8; ++i) { r[i] = lo[i]; r[i + 8] = hi[i]; }
  return r;
}

__device__ __forceinline__ float dot64(const float (&a)[DDIM], const float (&b)[DDIM]) {
  float s = 0.0f;
#pragma unroll
  for (int i = 0; i < DDIM; ++i) s = fmaf(a[i], b[i], s);
  return s;
}

__device__ __forceinline__ void mobius_add(const float (&x)[DDIM], const float (&y)[DDIM],
                                           float (&o)[DDIM]) {
  float xy = dot64(x, y), x2 = dot64(x, x), y2 = dot64(y, y);
  float ca  = 1.0f + 2.0f * xy + y2;
  float cb  = 1.0f - x2;
  float den = 1.0f + 2.0f * xy + x2 * y2;
  float inv = 1.0f / fmaxf(den, HEPS);
#pragma unroll
  for (int i = 0; i < DDIM; ++i) o[i] = (ca * x[i] + cb * y[i]) * inv;
}

__device__ __forceinline__ void exp_map(const float (&x)[DDIM], const float (&v)[DDIM],
                                        float (&o)[DDIM]) {
  float vn  = fmaxf(sqrtf(dot64(v, v)), 1e-15f);
  float lam = 2.0f / fmaxf(1.0f - dot64(x, x), HEPS);
  float f   = fast_tanh(0.5f * lam * vn) / vn;
  float sec[DDIM];
#pragma unroll
  for (int i = 0; i < DDIM; ++i) sec[i] = f * v[i];
  mobius_add(x, sec, o);
}

// parallel transport x->y of v:  gyr[y, -x] v * lambda_x / lambda_y
__device__ __forceinline__ void ptransp(const float (&x)[DDIM], const float (&y)[DDIM],
                                        const float (&v)[DDIM], float (&o)[DDIM]) {
  float nx[DDIM];
#pragma unroll
  for (int i = 0; i < DDIM; ++i) nx[i] = -x[i];
  float t1[DDIM]; mobius_add(nx, v, t1);   // (-x) + v
  float t2[DDIM]; mobius_add(y, t1, t2);   // y + ((-x) + v)
  float t3[DDIM]; mobius_add(y, nx, t3);   // y + (-x)
  float nt3[DDIM];
#pragma unroll
  for (int i = 0; i < DDIM; ++i) nt3[i] = -t3[i];
  float t4[DDIM]; mobius_add(nt3, t2, t4);
  float lx = 2.0f / fmaxf(1.0f - dot64(x, x), HEPS);
  float ly = 2.0f / fmaxf(1.0f - dot64(y, y), HEPS);
  float sc = lx / ly;
#pragma unroll
  for (int i = 0; i < DDIM; ++i) o[i] = t4[i] * sc;
}

// grad_q V(q) for the 32 rows owned by this wave. x = this lane's row.
// Forward: h = x@W1+b1 (K=D=64, two chained 16x16x32 WMMAs per 16-col tile)
// Backward: g = ((1-tanh^2(h))*w2) @ W1^T  (K=H=512, accumulated over 16 chunks)
__device__ __forceinline__ void grad_potential(
    const float (&x)[DDIM], float (&g)[DDIM],
    const _Float16* __restrict__ w1f, const _Float16* __restrict__ w1b,
    const float* __restrict__ b1s, const float* __restrict__ w2s,
    _Float16* stgh, float* stgf, int lane)
{
  const int r16   = lane & 15;
  const int kb    = (lane & 16) ? 8 : 0;   // K-offset of this lane-half (A/B layout)
  const int rhalf = (lane & 16) ? 8 : 0;   // M-offset of this lane-half (C layout)

  __syncthreads();
  // stage this lane's row as f16, row-major [32][64]
#pragma unroll
  for (int c = 0; c < 8; ++c) {
    v8h t;
#pragma unroll
    for (int j = 0; j < 8; ++j) t[j] = (_Float16)x[c * 8 + j];
    *(v8h*)(stgh + lane * 64 + c * 8) = t;
  }
  __syncthreads();

  // A fragments: tile t = rows 16t..16t+15, kc = K-chunk of D (0..31, 32..63)
  v16h aq[2][2];
#pragma unroll
  for (int t = 0; t < 2; ++t)
#pragma unroll
    for (int kc = 0; kc < 2; ++kc)
      aq[t][kc] = ld16s(stgh + (t * 16 + r16) * 64 + kc * 32 + kb);
  __syncthreads();

  v8f gacc[2][4];
#pragma unroll
  for (int t = 0; t < 2; ++t)
#pragma unroll
    for (int nd = 0; nd < 4; ++nd)
#pragma unroll
      for (int r = 0; r < 8; ++r) gacc[t][nd][r] = 0.0f;

  for (int nc = 0; nc < 16; ++nc) {   // 16 chunks of 32 columns of H
    // ---- forward WMMAs: h tiles for columns [nc*32, nc*32+32) ----
    v8f hT[2][2];
#pragma unroll
    for (int s = 0; s < 2; ++s) {
      const int nt = nc * 2 + s;
      v16h b0  = ld16c(w1f + ((size_t)(nt)       * 32 + lane) * 16);  // kc=0
      v16h b1v = ld16c(w1f + ((size_t)(32 + nt)  * 32 + lane) * 16);  // kc=1
#pragma unroll
      for (int t = 0; t < 2; ++t) {
        v8f c;
#pragma unroll
        for (int r = 0; r < 8; ++r) c[r] = 0.0f;
        c = __builtin_amdgcn_wmma_f32_16x16x32_f16(false, aq[t][0], false, b0,
                                                   (short)0, c, false, false);
        c = __builtin_amdgcn_wmma_f32_16x16x32_f16(false, aq[t][1], false, b1v,
                                                   (short)0, c, false, false);
        hT[t][s] = c;
      }
    }
    // ---- activation: u = (1 - tanh^2(h + b1)) * w2; store f16 to staging ----
#pragma unroll
    for (int s = 0; s < 2; ++s) {
      const int col = nc * 32 + s * 16 + r16;
      const float bb = b1s[col];
      const float ww = w2s[col];
#pragma unroll
      for (int t = 0; t < 2; ++t)
#pragma unroll
        for (int r = 0; r < 8; ++r) {
          float th = fast_tanh(hT[t][s][r] + bb);
          float u  = (1.0f - th * th) * ww;
          stgh[(t * 16 + rhalf + r) * 32 + s * 16 + r16] = (_Float16)u;
        }
    }
    __syncthreads();
    // ---- re-layout u (C -> A) and do backward WMMAs (K = this H chunk) ----
    v16h au[2];
#pragma unroll
    for (int t = 0; t < 2; ++t)
      au[t] = ld16s(stgh + (t * 16 + r16) * 32 + kb);
    __syncthreads();
#pragma unroll
    for (int t = 0; t < 2; ++t)
#pragma unroll
      for (int nd = 0; nd < 4; ++nd) {
        v16h bb = ld16c(w1b + ((size_t)(nc * 4 + nd) * 32 + lane) * 16);
        gacc[t][nd] = __builtin_amdgcn_wmma_f32_16x16x32_f16(
            false, au[t], false, bb, (short)0, gacc[t][nd], false, false);
      }
  }

  // ---- writeback gradient: C-layout tiles -> per-lane rows via LDS ----
#pragma unroll
  for (int t = 0; t < 2; ++t)
#pragma unroll
    for (int nd = 0; nd < 4; ++nd)
#pragma unroll
      for (int r = 0; r < 8; ++r)
        stgf[(t * 16 + rhalf + r) * 64 + nd * 16 + r16] = gacc[t][nd][r];
  __syncthreads();
#pragma unroll
  for (int i = 0; i < 16; ++i) {
    v4f v = *(const v4f*)(stgf + lane * 64 + i * 4);
    g[i * 4 + 0] = v[0]; g[i * 4 + 1] = v[1];
    g[i * 4 + 2] = v[2]; g[i * 4 + 3] = v[3];
  }
}

__global__ __launch_bounds__(THREADS)
void hyper_leapfrog_kernel(const float* __restrict__ qin, const float* __restrict__ pin,
                           const float* __restrict__ W1, const float* __restrict__ b1g,
                           const float* __restrict__ W2g,
                           float* __restrict__ qout, float* __restrict__ pout, int B)
{
  extern __shared__ __align__(16) char smem[];
  _Float16* w1f = (_Float16*)(smem + OFF_W1F);
  _Float16* w1b = (_Float16*)(smem + OFF_W1B);
  float*    b1s = (float*)(smem + OFF_B1);
  float*    w2s = (float*)(smem + OFF_W2);
  const int tid  = threadIdx.x;
  const int lane = tid & 31;
  const int wave = tid >> 5;
  _Float16* stgh = (_Float16*)(smem + OFF_STG + wave * STG_PER_WAVE);
  float*    stgf = (float*)   (smem + OFF_STG + wave * STG_PER_WAVE);

  long row = (long)blockIdx.x * THREADS + tid;
  if (row >= B) row = B - 1;   // uniform-safe clamp (B is a multiple of 256)
  __builtin_prefetch(qin + row * DDIM, 0, 0);   // global_prefetch_b8
  __builtin_prefetch(pin + row * DDIM, 0, 0);

  for (int i = tid; i < HDIM; i += THREADS) { b1s[i] = b1g[i]; w2s[i] = W2g[i]; }

  // --- stage W1 into LDS via TDM (2 chunks of 32 rows), then pre-swizzle ---
  // into per-lane WMMA B-fragment slabs:
  //  fwd frag (kc, nt): element(k = kc*32+koffs, n = nt*16+n16) = W1[k][n]
  //  bwd frag (kh, nd): element(k = kh*32+koffs, n = nd*16+n16) = W1[n][k]
  const float* rawW = (const float*)(smem + OFF_STG);   // 32x512 f32 chunk
#pragma unroll 1
  for (int ph = 0; ph < 2; ++ph) {
    __syncthreads();      // previous phase's gathers done before overwrite
#ifdef HAVE_TDM
    if (wave == 0) {
      tdm_load_w1_chunk(W1 + (size_t)ph * 32 * HDIM, (unsigned)OFF_STG);
      __builtin_amdgcn_s_wait_tensorcnt(0);
    }
#else
    for (int i = tid * 4; i < 32 * HDIM; i += THREADS * 4)
      *(v4f*)((float*)(smem + OFF_STG) + i) =
          *(const v4f*)(W1 + (size_t)ph * 32 * HDIM + i);
#endif
    __syncthreads();      // chunk visible to all waves
    for (int slot = tid; slot < 2048; slot += THREADS) {
      const int frag = slot >> 5;          // 0..63
      const int ln   = slot & 31;
      const int kb0  = (ln & 16) ? 8 : 0;
      const int c16  = ln & 15;
      if (frag < 32) {                     // forward frags, kc = ph, nt = frag
        _Float16* dst = w1f + ((size_t)(ph * 32 + frag) * 32 + ln) * 16;
#pragma unroll
        for (int i = 0; i < 16; ++i) {
          const int kl = kb0 + ((i < 8) ? i : (i + 8));     // local row 0..31
          dst[i] = (_Float16)rawW[kl * HDIM + frag * 16 + c16];
        }
      } else {                             // backward frags, nd in {2ph, 2ph+1}
        const int f2  = frag - 32;         // 0..31
        const int kh  = f2 >> 1;           // 0..15
        const int ndl = f2 & 1;
        const int nd  = ph * 2 + ndl;
        const int ddl = ndl * 16 + c16;    // local row 0..31
        _Float16* dst = w1b + ((size_t)(kh * 4 + nd) * 32 + ln) * 16;
#pragma unroll
        for (int i = 0; i < 16; ++i) {
          const int k = kh * 32 + kb0 + ((i < 8) ? i : (i + 8));
          dst[i] = (_Float16)rawW[ddl * HDIM + k];
        }
      }
    }
  }
  __syncthreads();

  float q[DDIM], p[DDIM];
#pragma unroll
  for (int i = 0; i < 16; ++i) {
    v4f tq = *(const v4f*)(qin + row * DDIM + i * 4);
    v4f tp = *(const v4f*)(pin + row * DDIM + i * 4);
    q[i * 4 + 0] = tq[0]; q[i * 4 + 1] = tq[1]; q[i * 4 + 2] = tq[2]; q[i * 4 + 3] = tq[3];
    p[i * 4 + 0] = tp[0]; p[i * 4 + 1] = tp[1]; p[i * 4 + 2] = tp[2]; p[i * 4 + 3] = tp[3];
  }

#pragma unroll 1
  for (int step = 0; step < HSTEPS; ++step) {
    float gv[DDIM];
    grad_potential(q, gv, w1f, w1b, b1s, w2s, stgh, stgf, lane);

    // grad_q H(q,p) = gradV + grad_q T ; T = 0.5||p||^2 * clip(1-||q||^2)^2 / 4
    const float sq    = dot64(q, q);
    const float pm    = dot64(p, p);
    const float onems = 1.0f - sq;
    const float kcoef = (onems > HEPS) ? (-0.5f * pm * onems) : 0.0f;
    float ph2[DDIM];
#pragma unroll
    for (int i = 0; i < DDIM; ++i) ph2[i] = p[i] - 0.5f * HDT * (gv[i] + kcoef * q[i]);

    // gHp = p_half / lambda(q)^2
    const float clipv = fmaxf(onems, HEPS);
    const float invl2 = 0.25f * clipv * clipv;
    float vv[DDIM];
#pragma unroll
    for (int i = 0; i < DDIM; ++i) vv[i] = HDT * ph2[i] * invl2;

    float qn[DDIM];
    exp_map(q, vv, qn);
    float pt[DDIM];
    ptransp(q, qn, ph2, pt);

    grad_potential(qn, gv, w1f, w1b, b1s, w2s, stgh, stgf, lane);
    const float sqn    = dot64(qn, qn);
    const float pm2    = dot64(pt, pt);
    const float onems2 = 1.0f - sqn;
    const float kcoef2 = (onems2 > HEPS) ? (-0.5f * pm2 * onems2) : 0.0f;
#pragma unroll
    for (int i = 0; i < DDIM; ++i) {
      p[i] = pt[i] - 0.5f * HDT * (gv[i] + kcoef2 * qn[i]);
      q[i] = qn[i];
    }
  }

#pragma unroll
  for (int i = 0; i < 16; ++i) {
    v4f tq = { q[i * 4 + 0], q[i * 4 + 1], q[i * 4 + 2], q[i * 4 + 3] };
    v4f tp = { p[i * 4 + 0], p[i * 4 + 1], p[i * 4 + 2], p[i * 4 + 3] };
    *(v4f*)(qout + row * DDIM + i * 4) = tq;
    *(v4f*)(pout + row * DDIM + i * 4) = tp;
  }
}

extern "C" void kernel_launch(void* const* d_in, const int* in_sizes, int n_in,
                              void* d_out, int out_size, void* d_ws, size_t ws_size,
                              hipStream_t stream) {
  const float* q  = (const float*)d_in[0];
  const float* p  = (const float*)d_in[1];
  const float* W1 = (const float*)d_in[2];
  const float* b1 = (const float*)d_in[3];
  const float* W2 = (const float*)d_in[4];
  (void)d_ws; (void)ws_size; (void)n_in; (void)out_size;

  const int B = in_sizes[0] / DDIM;
  float* qout = (float*)d_out;
  float* pout = qout + (size_t)B * DDIM;

  const int blocks = (B + THREADS - 1) / THREADS;
  (void)hipFuncSetAttribute((const void*)hyper_leapfrog_kernel,
                            hipFuncAttributeMaxDynamicSharedMemorySize, SMEM_BYTES);
  hipLaunchKernelGGL(hyper_leapfrog_kernel, dim3(blocks), dim3(THREADS),
                     SMEM_BYTES, stream, q, p, W1, b1, W2, qout, pout, B);
}
